// CRMF_35296041239144
// MI455X (gfx1250) — matrix operand order repaired and checked by reading, compiled, not run on hardware
//
#include <hip/hip_runtime.h>
#include <hip/hip_bf16.h>

typedef __attribute__((ext_vector_type(16))) _Float16 v16h;
typedef __attribute__((ext_vector_type(8)))  _Float16 v8h;
typedef __attribute__((ext_vector_type(8)))  float    v8f;

#define OBS_LEN 12
#define KSAMP   20
#define BATCH   2048
#define HDIM    128
#define MIDDIM  256
#define NROWS   (KSAMP * BATCH)   // 40960
#define ROWS_PER_BLOCK 128

// ---------- device helpers ----------

__device__ __forceinline__ float hsig(float x) {
    return fminf(fmaxf(x * 0.16666667f + 0.5f, 0.0f), 1.0f);
}
__device__ __forceinline__ float clip1(float x) {
    return fminf(fmaxf(x, -1.0f), 1.0f);
}

// Async global->LDS copy, 16B per lane, tracked by ASYNCcnt (CDNA5).
// VDST = per-lane LDS byte address, VADDR = 64-bit global address.
__device__ __forceinline__ void async_copy_b128(unsigned lds_addr, const void* g) {
    asm volatile("global_load_async_to_lds_b128 %0, %1, off"
                 :: "v"(lds_addr), "v"(g) : "memory");
}
__device__ __forceinline__ void wait_async0() {
    asm volatile("s_wait_asynccnt 0x0" ::: "memory");
}

// A-fragment (16x32 f16, M x K). rowp points at this lane's row (m = lane&15).
// lane<16 halves = K {k0+0..15}; lane>=16: {k0+16..31} (contiguous, matches B)
__device__ __forceinline__ v16h load_a_row(const _Float16* rowp, int k0, int hi) {
    v8h c0 = *(const v8h*)(rowp + k0 + hi * 16);
    v8h c1 = *(const v8h*)(rowp + k0 + hi * 16 + 8);
    v16h r;
#pragma unroll
    for (int i = 0; i < 8; ++i) { r[i] = c0[i]; r[8 + i] = c1[i]; }
    return r;
}

// B-fragment (32x16 f16, K x N) from transposed weights BT[n][k] (ld = ldk).
// lane&15 = N, hi selects K 0-15 vs 16-31: one contiguous 32B load per lane.
__device__ __forceinline__ v16h load_bT(const _Float16* bt, int ldk, int n0,
                                        int llo, int k0, int hi) {
    return *(const v16h*)(bt + (size_t)(n0 + llo) * ldk + k0 + hi * 16);
}

// ---------- kernel 1: convert weights to f16, B-friendly layouts ----------
// ws halves: [0,32768)      w0T [256][128]  w0T[n][k] = map_w0[k][n]
//            [32768,65536)  w1T [128][256]  w1T[n][k] = map_w1[k][n]
//            [65536,131072) whh [512][128]  direct f32->f16
__global__ void CRMF_convert_weights(const float* __restrict__ w0,
                                     const float* __restrict__ w1,
                                     const float* __restrict__ whh,
                                     _Float16* __restrict__ ws) {
    int i = blockIdx.x * 256 + threadIdx.x;
    if (i < 32768) {
        int n = i >> 7, k = i & 127;
        ws[i] = (_Float16)w0[k * 256 + n];
    } else if (i < 65536) {
        int l = i - 32768;
        int n = l >> 8, k = l & 255;
        ws[i] = (_Float16)w1[k * 128 + n];
    } else if (i < 131072) {
        ws[i] = (_Float16)whh[i - 65536];
    }
}

// Issue async stage of one 16KB w_hh j-tile panel (4 gates x 16 rows x 128 k).
__device__ __forceinline__ void stage_panel_async(
        _Float16 (*pbuf)[4][16][128], int buf, int tt,
        const _Float16* __restrict__ whhT, int tid) {
#pragma unroll
    for (int k = 0; k < 4; ++k) {
        int u = tid + k * 256;          // 1024 x 16B units
        int rowi = u >> 4;
        int off = (u & 15) * 8;
        int g = rowi >> 4, nn = rowi & 15;
        unsigned lds = (unsigned)(uintptr_t)&pbuf[buf][g][nn][off];
        const _Float16* src = whhT + ((size_t)(g * 128 + tt * 16 + nn) * 128 + off);
        async_copy_b128(lds, src);
    }
}

// ---------- kernel 2: persistent row-tile MLP + LSTM ----------
__global__ __launch_bounds__(256) void CRMF_lstm_kernel(
    const float* __restrict__ obs,       // [12][2048][3]
    const float* __restrict__ hidden,    // [40960][128]
    const float* __restrict__ map_b0,    // [256]
    const float* __restrict__ map_b1,    // [128]
    const float* __restrict__ w_ih,      // [512][2]
    const float* __restrict__ b_ih,      // [512]
    const float* __restrict__ b_hh,      // [512]
    const float* __restrict__ out_w,     // [128][2]
    const float* __restrict__ out_b,     // [2]
    const _Float16* __restrict__ w0T,    // [256][128]
    const _Float16* __restrict__ w1T,    // [128][256]
    const _Float16* __restrict__ whhT,   // [512][128]
    float* __restrict__ out)             // [12][40960][2]
{
    __shared__ __align__(32) _Float16 hbuf[128][128];        // 32 KB: h state (f16)
    __shared__ __align__(32) _Float16 pbuf[2][4][16][128];   // 32 KB: dbl-buffered panels
    // phase-1 transpose bounce aliases panel buffer 0 (disjoint lifetime)
    _Float16 (*midt)[32] = reinterpret_cast<_Float16(*)[32]>(&pbuf[0][0][0][0]);

    const int tid  = threadIdx.x;
    const int wave = tid >> 5;
    const int lane = tid & 31;
    const int llo  = lane & 15;
    const int hi   = lane >> 4;
    const int wrow = wave * 16;
    const int rowBase = blockIdx.x * ROWS_PER_BLOCK;

    // ---- phase 0: stage initial hidden rows into hbuf as f16 ----
    for (int i = tid; i < (128 * 128) / 4; i += 256) {
        float4 v = ((const float4*)hidden)[(size_t)rowBase * 32 + i];
        int r = i >> 5, c4 = (i & 31) * 4;
        hbuf[r][c4 + 0] = (_Float16)v.x;
        hbuf[r][c4 + 1] = (_Float16)v.y;
        hbuf[r][c4 + 2] = (_Float16)v.z;
        hbuf[r][c4 + 3] = (_Float16)v.w;
    }
    __syncthreads();

    // ---- phase 1: mapping MLP  h0 = leakyrelu(h@w0+b0)@w1 + b1 ----
    {
        v16h a[4];
#pragma unroll
        for (int kt = 0; kt < 4; ++kt)
            a[kt] = load_a_row(&hbuf[wrow + llo][0], kt * 32, hi);

        const v8f vz = {};
        v8f h0acc[8];
#pragma unroll
        for (int nt = 0; nt < 8; ++nt) h0acc[nt] = vz;

#pragma unroll 2
        for (int kc = 0; kc < 8; ++kc) {          // 32-wide chunk of MID
            v8f m0 = vz, m1 = vz;
#pragma unroll
            for (int kt = 0; kt < 4; ++kt) {      // K = 128 over hidden
                v16h bf0 = load_bT(w0T, 128, kc * 32,      llo, kt * 32, hi);
                v16h bf1 = load_bT(w0T, 128, kc * 32 + 16, llo, kt * 32, hi);
                m0 = __builtin_amdgcn_wmma_f32_16x16x32_f16(false, a[kt], false, bf0,
                                                            (short)0, m0, false, false);
                m1 = __builtin_amdgcn_wmma_f32_16x16x32_f16(false, a[kt], false, bf1,
                                                            (short)0, m1, false, false);
            }
            float bb0 = map_b0[kc * 32 + llo];
            float bb1 = map_b0[kc * 32 + 16 + llo];
#pragma unroll
            for (int e = 0; e < 8; ++e) {
                int m = e + hi * 8;
                float v0 = m0[e] + bb0;  v0 = v0 > 0.f ? v0 : 0.01f * v0;
                float v1 = m1[e] + bb1;  v1 = v1 > 0.f ? v1 : 0.01f * v1;
                midt[wrow + m][llo]      = (_Float16)v0;
                midt[wrow + m][llo + 16] = (_Float16)v1;
            }
            v16h am = load_a_row(&midt[wrow + llo][0], 0, hi);
#pragma unroll
            for (int nt = 0; nt < 8; ++nt) {
                v16h bw = load_bT(w1T, 256, nt * 16, llo, kc * 32, hi);
                h0acc[nt] = __builtin_amdgcn_wmma_f32_16x16x32_f16(false, am, false, bw,
                                                                   (short)0, h0acc[nt],
                                                                   false, false);
            }
        }
#pragma unroll
        for (int nt = 0; nt < 8; ++nt) {
            float bb = map_b1[nt * 16 + llo];
#pragma unroll
            for (int e = 0; e < 8; ++e) {
                int m = e + hi * 8;
                hbuf[wrow + m][nt * 16 + llo] = (_Float16)(h0acc[nt][e] + bb);
            }
        }
    }

    // ---- phase 2: 12 LSTM steps; h in LDS (f16), c in VGPRs ----
    float cst[8][8];
#pragma unroll
    for (int tt = 0; tt < 8; ++tt)
#pragma unroll
        for (int e = 0; e < 8; ++e) cst[tt][e] = 0.0f;

    for (int step = 0; step < OBS_LEN; ++step) {
        const int te = step ? step - 1 : 0;

        // per-lane x inputs for the 8 rows this lane touches (m = e + 8*hi)
        float xr0[8], xr1[8];
#pragma unroll
        for (int e = 0; e < 8; ++e) {
            int b = (rowBase + wrow + hi * 8 + e) & (BATCH - 1);
            const float* xp = obs + ((size_t)te * BATCH + b) * 3;
            xr0[e] = xp[0]; xr1[e] = xp[1];
        }
        if (step < OBS_LEN - 1)  // warm L2 for next step's inputs
            __builtin_prefetch(obs + ((size_t)step * BATCH +
                               (rowBase & (BATCH - 1))) * 3, 0, 1);

        __syncthreads();                       // h writes visible, panels free
        stage_panel_async(pbuf, 0, 0, whhT, tid);

        // snapshot this wave's h rows as A fragments while DMA runs
        v16h a[4];
#pragma unroll
        for (int kt = 0; kt < 4; ++kt)
            a[kt] = load_a_row(&hbuf[wrow + llo][0], kt * 32, hi);

        wait_async0();
        __syncthreads();

#pragma unroll
        for (int tt = 0; tt < 8; ++tt) {       // j-tile: gate cols tt*16..+15
            const int cur = tt & 1;
            if (tt < 7)                        // prefetch next panel via async DMA
                stage_panel_async(pbuf, cur ^ 1, tt + 1, whhT, tid);

            const v8f vz = {};
            v8f acc[4];
#pragma unroll
            for (int g = 0; g < 4; ++g) acc[g] = vz;
#pragma unroll
            for (int g = 0; g < 4; ++g)
#pragma unroll
                for (int kt = 0; kt < 4; ++kt) {
                    v16h bf = *(const v16h*)&pbuf[cur][g][llo][kt * 32 + hi * 16];
                    acc[g] = __builtin_amdgcn_wmma_f32_16x16x32_f16(false, a[kt], false, bf,
                                                                    (short)0, acc[g],
                                                                    false, false);
                }

            const int j = tt * 16 + llo;
            float bi  = b_ih[j]       + b_hh[j];
            float bf_ = b_ih[128 + j] + b_hh[128 + j];
            float bg  = b_ih[256 + j] + b_hh[256 + j];
            float bo  = b_ih[384 + j] + b_hh[384 + j];
            float wi0 = w_ih[j * 2],         wi1 = w_ih[j * 2 + 1];
            float wf0 = w_ih[(128 + j) * 2], wf1 = w_ih[(128 + j) * 2 + 1];
            float wg0 = w_ih[(256 + j) * 2], wg1 = w_ih[(256 + j) * 2 + 1];
            float wo0 = w_ih[(384 + j) * 2], wo1 = w_ih[(384 + j) * 2 + 1];
#pragma unroll
            for (int e = 0; e < 8; ++e) {
                int m = e + hi * 8;
                float iv = hsig (acc[0][e] + bi  + xr0[e] * wi0 + xr1[e] * wi1);
                float fv = hsig (acc[1][e] + bf_ + xr0[e] * wf0 + xr1[e] * wf1);
                float gv = clip1(acc[2][e] + bg  + xr0[e] * wg0 + xr1[e] * wg1);
                float ov = hsig (acc[3][e] + bo  + xr0[e] * wo0 + xr1[e] * wo1);
                float cn = fv * cst[tt][e] + iv * gv;
                cst[tt][e] = cn;
                hbuf[wrow + m][j] = (_Float16)(ov * clip1(cn));
            }

            wait_async0();                     // next panel landed in LDS
            __syncthreads();
        }

        // output projection: 32 lanes -> (row m = lane>>1, coord = lane&1)
        {
            int m = lane >> 1, cc = lane & 1;
            float s = out_b[cc];
#pragma unroll 8
            for (int jj = 0; jj < 128; ++jj)
                s += (float)hbuf[wrow + m][jj] * out_w[jj * 2 + cc];
            size_t row = (size_t)rowBase + wrow + m;
            out[((size_t)step * NROWS + row) * 2 + cc] = s;
        }
    }
}

extern "C" void kernel_launch(void* const* d_in, const int* in_sizes, int n_in,
                              void* d_out, int out_size, void* d_ws, size_t ws_size,
                              hipStream_t stream) {
    const float* obs     = (const float*)d_in[0];
    const float* hidden  = (const float*)d_in[1];
    const float* map_w0  = (const float*)d_in[2];
    const float* map_b0  = (const float*)d_in[3];
    const float* map_w1  = (const float*)d_in[4];
    const float* map_b1  = (const float*)d_in[5];
    const float* w_ih    = (const float*)d_in[6];
    const float* w_hh    = (const float*)d_in[7];
    const float* b_ih    = (const float*)d_in[8];
    const float* b_hh    = (const float*)d_in[9];
    const float* out_w   = (const float*)d_in[10];
    const float* out_b   = (const float*)d_in[11];

    _Float16* wsh = (_Float16*)d_ws;                // 256 KB of f16 weights
    CRMF_convert_weights<<<512, 256, 0, stream>>>(map_w0, map_w1, w_hh, wsh);

    CRMF_lstm_kernel<<<NROWS / ROWS_PER_BLOCK, 256, 0, stream>>>(
        obs, hidden, map_b0, map_b1, w_ih, b_ih, b_hh, out_w, out_b,
        wsh, wsh + 32768, wsh + 65536, (float*)d_out);
}
